// GATEncoder_48799418417430
// MI455X (gfx1250) — matrix operand order repaired and checked by reading, not compiled, not run on hardware
//
#include <hip/hip_runtime.h>
#include <hip/hip_bf16.h>

typedef __attribute__((ext_vector_type(16))) _Float16 v16h;
typedef __attribute__((ext_vector_type(8)))  _Float16 v8h;
typedef __attribute__((ext_vector_type(8)))  float    v8f;

#define N_NODES  50000      // multiple of 16 -> no M guards needed in GEMM
#define N_EDGES  800000
#define E_TOT    (N_EDGES + N_NODES)   // with self loops
#define N_GRAPHS 512
#define NEG_SLOPE 0.2f

// ---------- order-preserving float<->uint encoding for atomicMax softmax-max ----------
__device__ __forceinline__ unsigned enc_f32(float f) {
  unsigned u = __float_as_uint(f);
  return (u & 0x80000000u) ? ~u : (u | 0x80000000u);
}
__device__ __forceinline__ float dec_f32(unsigned e) {
  return __uint_as_float((e & 0x80000000u) ? (e ^ 0x80000000u) : ~e);
}
// enc(-inf) = ~0xFF800000
#define ENC_NEG_INF 0x007FFFFFu

__device__ __forceinline__ v16h mk16(v8h lo, v8h hi) {
  v16h r;
#pragma unroll
  for (int i = 0; i < 8; ++i) { r[i] = lo[i]; r[8 + i] = hi[i]; }
  return r;
}

// ---------------------------- fills ----------------------------
__global__ void k_fill_f32(float* __restrict__ p, float v, long n) {
  long i = blockIdx.x * (long)blockDim.x + threadIdx.x;
  if (i < n) p[i] = v;
}
__global__ void k_fill_u32(unsigned* __restrict__ p, unsigned v, long n) {
  long i = blockIdx.x * (long)blockDim.x + threadIdx.x;
  if (i < n) p[i] = v;
}

// ---------------- fp32 -> fp16 convert with zero padding (rows x cols) ----------------
__global__ void k_cvt_pad(const float* __restrict__ in, _Float16* __restrict__ out,
                          int nrows, int ncols, int colpad) {
  long i = blockIdx.x * (long)blockDim.x + threadIdx.x;
  long tot = (long)nrows * colpad;
  if (i >= tot) return;
  int r = (int)(i / colpad), c = (int)(i % colpad);
  float v = (c < ncols) ? in[(long)r * ncols + c] : 0.f;
  out[i] = (_Float16)v;
}
// same but also pads rows (for weights: [fo,fi] -> [fo_pad, fi_pad])
__global__ void k_cvt_pad2(const float* __restrict__ in, _Float16* __restrict__ out,
                           int nrows, int ncols, int rowpad, int colpad) {
  long i = blockIdx.x * (long)blockDim.x + threadIdx.x;
  long tot = (long)rowpad * colpad;
  if (i >= tot) return;
  int r = (int)(i / colpad), c = (int)(i % colpad);
  float v = (r < nrows && c < ncols) ? in[(long)r * ncols + c] : 0.f;
  out[i] = (_Float16)v;
}

// ---------------- WMMA GEMM: out[N,fo] = A16[N,fi_pad] @ W16[fo_pad,fi_pad]^T + bias ----
// One wave computes a 16(M) x 32(N) output tile: A fragment reused across 2 WMMAs.
// All inner-loop loads are unguarded 16B vector loads (buffers are zero-padded).
__global__ void k_gemm_wmma(const _Float16* __restrict__ A16,
                            const _Float16* __restrict__ W16, int fi_pad,
                            const float* __restrict__ bias, float* __restrict__ out,
                            int fo, int mtiles, int n2tiles) {
  int wave = (int)((blockIdx.x * (long)blockDim.x + threadIdx.x) >> 5);
  int lane = threadIdx.x & 31;
  if (wave >= mtiles * n2tiles) return;
  int mt = wave / n2tiles, nt = wave % n2tiles;
  int mbase = mt * 16, nbase = nt * 32;
  int half = lane >> 4, l15 = lane & 15;
  int m  = mbase + l15;        // A row held by this lane
  int n0 = nbase + l15;        // first output column held by this lane
  int n1 = n0 + 16;            // second output column

  const _Float16* ap  = A16 + (size_t)m  * fi_pad + half * 16;  // K-half 0/1 -> +16
  const _Float16* b0p = W16 + (size_t)n0 * fi_pad + half * 16;  // K-half 0/1 -> +16
  const _Float16* b1p = W16 + (size_t)n1 * fi_pad + half * 16;

  v8f c0 = {}, c1 = {};
  int ksteps = fi_pad >> 5;
  for (int ks = 0; ks < ksteps; ++ks) {
    v16h a  = mk16(*(const v8h*)ap,        *(const v8h*)(ap + 8));
    v16h b0 = mk16(*(const v8h*)b0p,       *(const v8h*)(b0p + 8));
    v16h b1 = mk16(*(const v8h*)b1p,       *(const v8h*)(b1p + 8));
    c0 = __builtin_amdgcn_wmma_f32_16x16x32_f16(false, a, false, b0,
                                                (short)0, c0, false, false);
    c1 = __builtin_amdgcn_wmma_f32_16x16x32_f16(false, a, false, b1,
                                                (short)0, c1, false, false);
    ap += 32; b0p += 32; b1p += 32;
  }
  // ---- store C/D: VGPR r -> row mbase + r + 8*half, cols n0 / n1 ----
  int rowb = mbase + 8 * half;
  if (n0 < fo) {
    float bv = bias[n0];
#pragma unroll
    for (int r = 0; r < 8; ++r) out[(size_t)(rowb + r) * fo + n0] = c0[r] + bv;
  }
  if (n1 < fo) {
    float bv = bias[n1];
#pragma unroll
    for (int r = 0; r < 8; ++r) out[(size_t)(rowb + r) * fo + n1] = c1[r] + bv;
  }
}

// ---------------- per-(edge,head) attention score + segment max ----------------
__global__ void k_score(const float* __restrict__ xl, const float* __restrict__ xr,
                        const int* __restrict__ srcv, const int* __restrict__ dstv,
                        const float* __restrict__ att, float* __restrict__ esc,
                        unsigned* __restrict__ nmax, int H, int C) {
  long i = blockIdx.x * (long)blockDim.x + threadIdx.x;
  long tot = (long)E_TOT * H;
  if (i >= tot) return;
  int h = (int)(i % H);
  long e = i / H;
  int s = (e < N_EDGES) ? srcv[e] : (int)(e - N_EDGES);
  int d = (e < N_EDGES) ? dstv[e] : (int)(e - N_EDGES);
  int fo = H * C;
  const float* pl = xl + (size_t)s * fo + h * C;
  const float* pr = xr + (size_t)d * fo + h * C;
  const float* pa = att + h * C;
  float acc = 0.f;
  for (int c4 = 0; c4 < C; c4 += 4) {
    float4 a = *(const float4*)(pl + c4);
    float4 b = *(const float4*)(pr + c4);
    float4 t = *(const float4*)(pa + c4);
    float m0 = a.x + b.x, m1 = a.y + b.y, m2 = a.z + b.z, m3 = a.w + b.w;
    m0 = m0 > 0.f ? m0 : NEG_SLOPE * m0;
    m1 = m1 > 0.f ? m1 : NEG_SLOPE * m1;
    m2 = m2 > 0.f ? m2 : NEG_SLOPE * m2;
    m3 = m3 > 0.f ? m3 : NEG_SLOPE * m3;
    acc += m0 * t.x + m1 * t.y + m2 * t.z + m3 * t.w;
  }
  esc[i] = acc;
  atomicMax(nmax + (size_t)d * H + h, enc_f32(acc));
}

// ---------------- p = exp(e - max[dst]); segment sum ----------------
__global__ void k_expsum(const int* __restrict__ dstv, float* __restrict__ esc,
                         const unsigned* __restrict__ nmax, float* __restrict__ denom,
                         int H) {
  long i = blockIdx.x * (long)blockDim.x + threadIdx.x;
  long tot = (long)E_TOT * H;
  if (i >= tot) return;
  int h = (int)(i % H);
  long e = i / H;
  int d = (e < N_EDGES) ? dstv[e] : (int)(e - N_EDGES);
  float mx = dec_f32(nmax[(size_t)d * H + h]);
  float p = __expf(esc[i] - mx);
  esc[i] = p;
  atomicAdd(denom + (size_t)d * H + h, p);
}

// ---------------- out[dst] += alpha * xl[src] ----------------
__global__ void k_aggregate(const float* __restrict__ xl,
                            const int* __restrict__ srcv, const int* __restrict__ dstv,
                            const float* __restrict__ esc, const float* __restrict__ denom,
                            float* __restrict__ hbuf, int H, int C) {
  long i = blockIdx.x * (long)blockDim.x + threadIdx.x;
  long tot = (long)E_TOT * H;
  if (i >= tot) return;
  int h = (int)(i % H);
  long e = i / H;
  int s = (e < N_EDGES) ? srcv[e] : (int)(e - N_EDGES);
  int d = (e < N_EDGES) ? dstv[e] : (int)(e - N_EDGES);
  int fo = H * C;
  float alpha = esc[i] / denom[(size_t)d * H + h];
  const float* pl = xl + (size_t)s * fo + h * C;
  float* po = hbuf + (size_t)d * fo + h * C;
  for (int c = 0; c < C; ++c) atomicAdd(po + c, pl[c] * alpha);
}

// ---------------- bias + ReLU + convert to f16 for next layer ----------------
__global__ void k_finalize_mid(const float* __restrict__ hbuf, const float* __restrict__ bias,
                               _Float16* __restrict__ x16, int fo) {
  long i = blockIdx.x * (long)blockDim.x + threadIdx.x;
  long tot = (long)N_NODES * fo;
  if (i >= tot) return;
  int j = (int)(i % fo);
  float v = hbuf[i] + bias[j];
  v = v > 0.f ? v : 0.f;
  x16[i] = (_Float16)v;   // fo == fi_pad (192) for layers 1..3
}
__global__ void k_finalize_last(const float* __restrict__ hbuf, const float* __restrict__ bias,
                                float* __restrict__ hfin) {
  long i = blockIdx.x * (long)blockDim.x + threadIdx.x;
  long tot = (long)N_NODES * 20;
  if (i >= tot) return;
  int j = (int)(i % 20);
  hfin[i] = hbuf[i] + bias[j];
}

// ---------------- global_add_pool over batch ----------------
__global__ void k_pool(const float* __restrict__ hfin, const int* __restrict__ batch,
                       float* __restrict__ out) {
  long i = blockIdx.x * (long)blockDim.x + threadIdx.x;
  long tot = (long)N_NODES * 20;
  if (i >= tot) return;
  int n = (int)(i / 20), j = (int)(i % 20);
  atomicAdd(out + (size_t)batch[n] * 20 + j, hfin[i]);
}

// =============================== host launch ===============================
static inline int blks(long n, int t) { return (int)((n + t - 1) / t); }

extern "C" void kernel_launch(void* const* d_in, const int* in_sizes, int n_in,
                              void* d_out, int out_size, void* d_ws, size_t ws_size,
                              hipStream_t stream) {
  const float* x    = (const float*)d_in[0];
  const int*   ei   = (const int*)d_in[1];
  const int*   batch= (const int*)d_in[2];
  const int*   srcv = ei;
  const int*   dstv = ei + N_EDGES;

  // workspace layout
  char* ws = (char*)d_ws;
  size_t off = 0;
  auto wsalloc = [&](size_t bytes) -> void* {
    void* p = ws + off;
    off = (off + bytes + 255) & ~(size_t)255;
    return p;
  };
  _Float16* x16  = (_Float16*)wsalloc((size_t)N_NODES * 192 * 2);
  float*    xl   = (float*)   wsalloc((size_t)N_NODES * 192 * 4);
  float*    xr   = (float*)   wsalloc((size_t)N_NODES * 192 * 4);
  float*    hbuf = (float*)   wsalloc((size_t)N_NODES * 192 * 4);
  float*    esc  = (float*)   wsalloc((size_t)E_TOT   * 3   * 4);
  unsigned* nmax = (unsigned*)wsalloc((size_t)N_NODES * 3   * 4);
  float*    denom= (float*)   wsalloc((size_t)N_NODES * 3   * 4);
  _Float16* wl16 = (_Float16*)wsalloc((size_t)192 * 192 * 2);   // padded f16 weights
  _Float16* wr16 = (_Float16*)wsalloc((size_t)192 * 192 * 2);
  float*    hfin = xr;  // reuse xr for final [N,20] activations

  const int FI[4] = {11, 192, 192, 192};
  const int CH[4] = {64, 64, 64, 20};
  const int HH[4] = {3, 3, 3, 1};

  // layer 0 input conversion (pad K 11 -> 32)
  k_cvt_pad<<<blks((long)N_NODES * 32, 256), 256, 0, stream>>>(x, x16, N_NODES, 11, 32);

  for (int l = 0; l < 4; ++l) {
    int fi = FI[l], C = CH[l], H = HH[l], fo = H * C;
    int fi_pad = (fi + 31) & ~31;     // 32, 192, 192, 192
    int fo_pad = (fo + 31) & ~31;     // 192, 192, 192, 32
    const float* Wl = (const float*)d_in[3 + 6 * l + 0];
    const float* bl = (const float*)d_in[3 + 6 * l + 1];
    const float* Wr = (const float*)d_in[3 + 6 * l + 2];
    const float* br = (const float*)d_in[3 + 6 * l + 3];
    const float* at = (const float*)d_in[3 + 6 * l + 4];
    const float* bb = (const float*)d_in[3 + 6 * l + 5];

    // --- pad+convert weights to f16 (tiny) ---
    long wtot = (long)fo_pad * fi_pad;
    k_cvt_pad2<<<blks(wtot, 256), 256, 0, stream>>>(Wl, wl16, fo, fi, fo_pad, fi_pad);
    k_cvt_pad2<<<blks(wtot, 256), 256, 0, stream>>>(Wr, wr16, fo, fi, fo_pad, fi_pad);

    // --- dense transforms via WMMA (16x32 tile per wave) ---
    int mtiles  = N_NODES / 16;       // 3125, exact
    int n2tiles = fo_pad / 32;        // 6,6,6,1
    long waves  = (long)mtiles * n2tiles;
    int gblocks = blks(waves * 32, 256);
    k_gemm_wmma<<<gblocks, 256, 0, stream>>>(x16, wl16, fi_pad, bl, xl, fo, mtiles, n2tiles);
    k_gemm_wmma<<<gblocks, 256, 0, stream>>>(x16, wr16, fi_pad, br, xr, fo, mtiles, n2tiles);

    // --- init segment buffers ---
    k_fill_u32<<<blks((long)N_NODES * H, 256), 256, 0, stream>>>(nmax, ENC_NEG_INF, (long)N_NODES * H);
    k_fill_f32<<<blks((long)N_NODES * H, 256), 256, 0, stream>>>(denom, 0.f, (long)N_NODES * H);
    k_fill_f32<<<blks((long)N_NODES * fo, 256), 256, 0, stream>>>(hbuf, 0.f, (long)N_NODES * fo);

    // --- edge attention ---
    long eh = (long)E_TOT * H;
    k_score    <<<blks(eh, 256), 256, 0, stream>>>(xl, xr, srcv, dstv, at, esc, nmax, H, C);
    k_expsum   <<<blks(eh, 256), 256, 0, stream>>>(dstv, esc, nmax, denom, H);
    k_aggregate<<<blks(eh, 256), 256, 0, stream>>>(xl, srcv, dstv, esc, denom, hbuf, H, C);

    // --- bias (+ReLU) / convert for next layer ---
    if (l < 3) {
      k_finalize_mid<<<blks((long)N_NODES * fo, 256), 256, 0, stream>>>(hbuf, bb, x16, fo);
    } else {
      k_finalize_last<<<blks((long)N_NODES * 20, 256), 256, 0, stream>>>(hbuf, bb, hfin);
    }
  }

  // --- global add pool ---
  k_fill_f32<<<blks((long)N_GRAPHS * 20, 256), 256, 0, stream>>>((float*)d_out, 0.f, (long)N_GRAPHS * 20);
  k_pool<<<blks((long)N_NODES * 20, 256), 256, 0, stream>>>(hfin, batch, (float*)d_out);
}
